// EquiGraspSO3DeformableAttn_3410204033230
// MI455X (gfx1250) — matrix operand order repaired and checked by reading, not compiled, not run on hardware
//
#include <hip/hip_runtime.h>
#include <hip/hip_bf16.h>
#include <math.h>

// ---------------------------------------------------------------------------
// EquiGrasp SO3 deformable attention, CDNA5 (gfx1250) wave32 + WMMA bf16.
//
// Algebraic refactor:
//   sim[q,g] = qk[q] . sf[q,g] + q[q] . (bk + pe_k[g]),  qk = q @ Wk
//   out_attn = (sum_g attn_g * sf[q,g]) @ Wv^T + bv      (since sum attn = 1)
// so the 409600-row K/V GEMMs collapse to M=16384 GEMMs and sf is never
// materialized in memory.
//
// Weights are pre-swizzled to fragment-ordered bf16 (no f32->bf16 in the WMMA
// hot loop), A operands are bf16, weights staged to LDS with CDNA5 async
// global->LDS loads (v4i-typed builtin signature per hipcc diagnostic).
// ---------------------------------------------------------------------------

typedef __attribute__((ext_vector_type(16))) __bf16 v16bf;
typedef __attribute__((ext_vector_type(8)))  __bf16 v8bf;
typedef __attribute__((ext_vector_type(8)))  float  v8f;
typedef unsigned short ushort_t;

typedef int v4i_t __attribute__((vector_size(16)));
typedef __attribute__((address_space(1))) v4i_t as1_v4i;
typedef __attribute__((address_space(3))) v4i_t as3_v4i;

#define C_DIM 128
#define HP    64
#define NCP   25
#define WPACK (C_DIM * C_DIM)   // elements in one packed weight (16384)

__device__ __forceinline__ unsigned short f2bfu(float f) {
    unsigned u = __builtin_bit_cast(unsigned, f);
    unsigned r = u + 0x7fffu + ((u >> 16) & 1u);     // round to nearest even
    return (unsigned short)(r >> 16);
}
__device__ __forceinline__ __bf16 f2bf(float f) {
    return __builtin_bit_cast(__bf16, f2bfu(f));
}

// ------------------------------ bilinear -----------------------------------
__device__ __forceinline__ float bilin(const float* __restrict__ pc, float X, float Y) {
    float u = fminf(fmaxf((X + 1.0f) * 0.5f * (float)(HP - 1), 0.0f), (float)(HP - 1));
    float v = fminf(fmaxf((Y + 1.0f) * 0.5f * (float)(HP - 1), 0.0f), (float)(HP - 1));
    float x0 = floorf(u), y0 = floorf(v);
    float wx = u - x0, wy = v - y0;
    int x0i = (int)x0, y0i = (int)y0;
    int x1i = x0i + 1 > HP - 1 ? HP - 1 : x0i + 1;
    int y1i = y0i + 1 > HP - 1 ? HP - 1 : y0i + 1;
    float f00 = pc[y0i * HP + x0i];
    float f01 = pc[y0i * HP + x1i];
    float f10 = pc[y1i * HP + x0i];
    float f11 = pc[y1i * HP + x1i];
    return f00 * (1.0f - wx) * (1.0f - wy) + f01 * wx * (1.0f - wy) +
           f10 * (1.0f - wx) * wy          + f11 * wx * wy;
}

__device__ __forceinline__ float tri_sample(const float* __restrict__ cxy,
                                            const float* __restrict__ cxz,
                                            const float* __restrict__ cyz,
                                            size_t chan_base,
                                            float x, float y, float z) {
    return bilin(cxy + chan_base, x, y) +
           bilin(cxz + chan_base, x, z) +
           bilin(cyz + chan_base, y, z);
}

// ------------------------- WMMA fragment helpers ---------------------------
// A (16x32 bf16) from bf16 row-major src.  ISA 7.12.2 16-bit A layout:
// lanes 0-15 -> M, elems 0..7 = K k0..k0+7 (+8 for hi half-wave),
// elems 8..15 = K k0+16..k0+23 (+8 for hi half-wave): two contiguous 8-runs.
__device__ __forceinline__ v16bf load_a_frag_bf(const __bf16* __restrict__ A, int ld,
                                                int lane, int k0) {
    int m  = lane & 15;
    int hi = (lane >> 4) & 1;
    const __bf16* row = A + (size_t)m * ld + k0 + hi * 8;
    v8bf lo = *(const v8bf*)(row);
    v8bf hi8 = *(const v8bf*)(row + 16);
    return __builtin_shufflevector(lo, hi8, 0, 1, 2, 3, 4, 5, 6, 7,
                                            8, 9, 10, 11, 12, 13, 14, 15);
}

// B packed fragment-ordered: [frag=kt*8+nt][lane][16] -> one 32B load per lane.
__device__ __forceinline__ void gemm16_p(const __bf16* __restrict__ A, int lda,
                                         const __bf16* __restrict__ Bp,
                                         v8f acc[8], int lane) {
#pragma unroll
    for (int kt = 0; kt < 4; ++kt) {
        int k0 = kt * 32;
        v16bf a = load_a_frag_bf(A, lda, lane, k0);
#pragma unroll
        for (int nt = 0; nt < 8; ++nt) {
            const v16bf* bp =
                (const v16bf*)(Bp + ((size_t)((kt << 3) + nt) * 32 + lane) * 16);
            acc[nt] = __builtin_amdgcn_wmma_f32_16x16x32_bf16(
                false, a, false, *bp, (short)0, acc[nt], false, false);
        }
    }
}

// ------------------- async global -> LDS weight staging --------------------
// Stage one packed weight (32 KB bf16) into LDS; 128 threads x 16 chunks x 16B.
__device__ __forceinline__ void stage_weight(const __bf16* __restrict__ g,
                                             __bf16* l, int tid) {
#if __has_builtin(__builtin_amdgcn_global_load_async_to_lds_b128)
#pragma unroll
    for (int i = 0; i < 16; ++i) {
        int chunk = tid + i * 128;
        __builtin_amdgcn_global_load_async_to_lds_b128(
            (as1_v4i*)(g + (size_t)chunk * 8),
            (as3_v4i*)(l + (size_t)chunk * 8),
            0, 0);
    }
#if __has_builtin(__builtin_amdgcn_s_wait_asynccnt)
    __builtin_amdgcn_s_wait_asynccnt(0);
#else
    asm volatile("s_wait_asynccnt 0x0" ::: "memory");
#endif
#else
#pragma unroll
    for (int i = 0; i < 16; ++i) {
        int chunk = tid + i * 128;
        *(uint4*)(l + (size_t)chunk * 8) = *(const uint4*)(g + (size_t)chunk * 8);
    }
#endif
}

// ------------------------ Kernel 0: weight packing -------------------------
// dst[frag][lane][j] = bf16( bt ? W[n*128+k] : W[k*128+n] )
// with k = kt*32 + (j&7) + (j>=8 ? 16 : 0) + (lane>=16 ? 8 : 0), n = nt*16 + (lane&15).
__global__ void pack_w_kernel(const float* __restrict__ W,
                              ushort_t* __restrict__ dst, int bt) {
    int t = blockIdx.x * blockDim.x + threadIdx.x;      // 0 .. 16383
    int j    = t & 15;
    int lane = (t >> 4) & 31;
    int frag = t >> 9;                                  // kt*8 + nt
    int kt = frag >> 3, nt = frag & 7;
    int hi = (lane >> 4) & 1;
    int n = nt * 16 + (lane & 15);
    int k = kt * 32 + (j & 7) + ((j >> 3) << 4) + (hi ? 8 : 0);
    float v = bt ? W[(size_t)n * C_DIM + k] : W[(size_t)k * C_DIM + n];
    dst[t] = f2bfu(v);
}

// ----------------------------- Kernel 1: geom ------------------------------
__global__ void geom_kernel(const float* __restrict__ qp,
                            const float* __restrict__ cxy,
                            const float* __restrict__ cxz,
                            const float* __restrict__ cyz,
                            const float* __restrict__ cp,
                            const float* __restrict__ Woff,
                            float* __restrict__ feat_ws,
                            ushort_t* __restrict__ feat_bf,
                            float* __restrict__ pts_ws,
                            int ns) {
    int qid = blockIdx.x;
    int c   = threadIdx.x;            // channel 0..127
    int b   = qid / ns;

    const float* qr = qp + (size_t)qid * 9;
    float px = qr[0], py = qr[1], pz = qr[2];
    float a1x = qr[3], a1y = qr[4], a1z = qr[5];
    float a2x = qr[6], a2y = qr[7], a2z = qr[8];

    float n1 = rsqrtf(a1x * a1x + a1y * a1y + a1z * a1z);
    float b1x = a1x * n1, b1y = a1y * n1, b1z = a1z * n1;
    float d   = b1x * a2x + b1y * a2y + b1z * a2z;
    float t2x = a2x - d * b1x, t2y = a2y - d * b1y, t2z = a2z - d * b1z;
    float n2 = rsqrtf(t2x * t2x + t2y * t2y + t2z * t2z);
    float b2x = t2x * n2, b2y = t2y * n2, b2z = t2z * n2;
    float b3x = b1y * b2z - b1z * b2y;
    float b3y = b1z * b2x - b1x * b2z;
    float b3z = b1x * b2y - b1y * b2x;

    size_t chan_base = ((size_t)(b * C_DIM + c)) * (HP * HP);
    float f = tri_sample(cxy, cxz, cyz, chan_base, px, py, pz);
    feat_ws[(size_t)qid * C_DIM + c] = f;
    feat_bf[(size_t)qid * C_DIM + c] = f2bfu(f);

    __shared__ float sfeat[C_DIM];
    __shared__ float soff[3];
    __shared__ float sR[9];
    sfeat[c] = f;
    if (c == 0) {
        sR[0] = b1x; sR[1] = b1y; sR[2] = b1z;
        sR[3] = b2x; sR[4] = b2y; sR[5] = b2z;
        sR[6] = b3x; sR[7] = b3y; sR[8] = b3z;
    }
    __syncthreads();

    if (c < 3) {
        float o = 0.0f;
#pragma unroll 8
        for (int i = 0; i < C_DIM; ++i) o += Woff[c * C_DIM + i] * sfeat[i];
        soff[c] = o;
    }
    __syncthreads();

    if (c < NCP * 3) {
        int g = c / 3, dd = c % 3;
        float cx = cp[g * 3 + 0], cy = cp[g * 3 + 1], cz = cp[g * 3 + 2];
        float base = (dd == 0) ? px : ((dd == 1) ? py : pz);
        float val = base + soff[dd] +
                    sR[dd * 3 + 0] * cx + sR[dd * 3 + 1] * cy + sR[dd * 3 + 2] * cz;
        pts_ws[((size_t)qid * NCP + g) * 3 + dd] = val;
    }
}

// ------------------------ Kernel 2: q and qk GEMMs -------------------------
__global__ void gemm_q_qk_kernel(const ushort_t* __restrict__ feat_bf_us,
                                 const ushort_t* __restrict__ Wqp_us,
                                 const ushort_t* __restrict__ Wkp_us,
                                 const float* __restrict__ bq,
                                 const float* __restrict__ peq,
                                 float* __restrict__ q_ws,
                                 float* __restrict__ qk_ws,
                                 int NQ) {
    __shared__ __align__(32) __bf16 wbuf[WPACK];           // 32 KB
    __shared__ __align__(32) __bf16 ytile[4][16 * C_DIM];  // 16 KB

    const __bf16* feat_bf = (const __bf16*)feat_bf_us;
    int tid  = threadIdx.x;
    int wave = tid >> 5;
    int lane = tid & 31;
    int m0 = (blockIdx.x * 4 + wave) * 16;
    bool active = (m0 < NQ);

    const float scale = sqrtf((float)C_DIM);
    int rowoff = ((lane >> 4) & 1) * 8;
    int ncol   = lane & 15;

    stage_weight((const __bf16*)Wqp_us, wbuf, tid);
    __syncthreads();

    v8f acc[8] = {};
    if (active) {
        gemm16_p(feat_bf + (size_t)m0 * C_DIM, C_DIM, wbuf, acc, lane);
#pragma unroll
        for (int nt = 0; nt < 8; ++nt) {
            int col = nt * 16 + ncol;
#pragma unroll
            for (int r = 0; r < 8; ++r) {
                int row = r + rowoff;
                float v = (acc[nt][r] + bq[col] + peq[col]) * scale;
                q_ws[(size_t)(m0 + row) * C_DIM + col] = v;
                ytile[wave][row * C_DIM + col] = f2bf(v);
            }
        }
    }
    __syncthreads();
    stage_weight((const __bf16*)Wkp_us, wbuf, tid);
    __syncthreads();

    v8f acc2[8] = {};
    if (active) {
        gemm16_p(&ytile[wave][0], C_DIM, wbuf, acc2, lane);
#pragma unroll
        for (int nt = 0; nt < 8; ++nt) {
            int col = nt * 16 + ncol;
#pragma unroll
            for (int r = 0; r < 8; ++r) {
                int row = r + rowoff;
                qk_ws[(size_t)(m0 + row) * C_DIM + col] = acc2[nt][r];
            }
        }
    }
}

// ------------------- Kernel 3: fused sample + attention --------------------
__global__ void attn_kernel(const float* __restrict__ pts,
                            const float* __restrict__ q_ws,
                            const float* __restrict__ qk_ws,
                            const float* __restrict__ bk,
                            const float* __restrict__ pek,
                            const float* __restrict__ cxy,
                            const float* __restrict__ cxz,
                            const float* __restrict__ cyz,
                            float* __restrict__ sbar,
                            ushort_t* __restrict__ sbar_bf,
                            int ns) {
    int qid = blockIdx.x;
    int c   = threadIdx.x;            // channel
    int b   = qid / ns;
    int wid  = c >> 5;
    int lane = c & 31;

    float qc  = q_ws [(size_t)qid * C_DIM + c];
    float qkc = qk_ws[(size_t)qid * C_DIM + c];
    float bkc = bk[c];
    size_t chan_base = ((size_t)(b * C_DIM + c)) * (HP * HP);

    __shared__ float swpart[4][32];
    __shared__ float ssim[32];

    float sf[NCP];
#pragma unroll 1
    for (int g = 0; g < NCP; ++g) {
        const float* pp = pts + ((size_t)qid * NCP + g) * 3;
        float X = pp[0], Y = pp[1], Z = pp[2];
        float v = tri_sample(cxy, cxz, cyz, chan_base, X, Y, Z);
        sf[g] = v;
        float r = v * qkc + qc * (bkc + pek[g * C_DIM + c]);
#pragma unroll
        for (int o = 16; o > 0; o >>= 1) r += __shfl_xor(r, o, 32);
        if (lane == 0) swpart[wid][g] = r;
    }
    __syncthreads();
    if (c < NCP) ssim[c] = swpart[0][c] + swpart[1][c] + swpart[2][c] + swpart[3][c];
    __syncthreads();

    float m = -3.0e38f;
#pragma unroll
    for (int g = 0; g < NCP; ++g) m = fmaxf(m, ssim[g]);
    float den = 0.0f, num = 0.0f;
#pragma unroll
    for (int g = 0; g < NCP; ++g) {
        float e = __expf(ssim[g] - m);
        den += e;
        num += e * sf[g];
    }
    float s = num / den;
    sbar   [(size_t)qid * C_DIM + c] = s;
    sbar_bf[(size_t)qid * C_DIM + c] = f2bfu(s);
}

// ------------------- Kernel 4: V and O projection GEMMs --------------------
__global__ void gemm_out_kernel(const ushort_t* __restrict__ sbar_bf_us,
                                const ushort_t* __restrict__ Wvp_us,
                                const ushort_t* __restrict__ Wop_us,
                                const float* __restrict__ bv,
                                const float* __restrict__ bo,
                                const float* __restrict__ feat,
                                float* __restrict__ out,
                                int NQ) {
    __shared__ __align__(32) __bf16 wbuf[WPACK];           // 32 KB
    __shared__ __align__(32) __bf16 ytile[4][16 * C_DIM];  // 16 KB

    const __bf16* sbar_bf = (const __bf16*)sbar_bf_us;
    int tid  = threadIdx.x;
    int wave = tid >> 5;
    int lane = tid & 31;
    int m0 = (blockIdx.x * 4 + wave) * 16;
    bool active = (m0 < NQ);

    int rowoff = ((lane >> 4) & 1) * 8;
    int ncol   = lane & 15;

    stage_weight((const __bf16*)Wvp_us, wbuf, tid);
    __syncthreads();

    v8f acc[8] = {};
    if (active) {
        gemm16_p(sbar_bf + (size_t)m0 * C_DIM, C_DIM, wbuf, acc, lane);
#pragma unroll
        for (int nt = 0; nt < 8; ++nt) {
            int col = nt * 16 + ncol;
#pragma unroll
            for (int r = 0; r < 8; ++r) {
                int row = r + rowoff;
                ytile[wave][row * C_DIM + col] = f2bf(acc[nt][r] + bv[col]);
            }
        }
    }
    __syncthreads();
    stage_weight((const __bf16*)Wop_us, wbuf, tid);
    __syncthreads();

    v8f acc2[8] = {};
    if (active) {
        gemm16_p(&ytile[wave][0], C_DIM, wbuf, acc2, lane);
#pragma unroll
        for (int nt = 0; nt < 8; ++nt) {
            int col = nt * 16 + ncol;
#pragma unroll
            for (int r = 0; r < 8; ++r) {
                int row = r + rowoff;
                size_t idx = (size_t)(m0 + row) * C_DIM + col;
                out[idx] = acc2[nt][r] + bo[col] + feat[idx];
            }
        }
    }
}

// --------------------------------- launch ----------------------------------
extern "C" void kernel_launch(void* const* d_in, const int* in_sizes, int n_in,
                              void* d_out, int out_size, void* d_ws, size_t ws_size,
                              hipStream_t stream) {
    const float* query_pos = (const float*)d_in[0];
    const float* c_xy      = (const float*)d_in[1];
    const float* c_xz      = (const float*)d_in[2];
    const float* c_yz      = (const float*)d_in[3];
    const float* Wq        = (const float*)d_in[4];
    const float* bq        = (const float*)d_in[5];
    const float* Wk        = (const float*)d_in[6];
    const float* bk        = (const float*)d_in[7];
    const float* Wv        = (const float*)d_in[8];
    const float* bv        = (const float*)d_in[9];
    const float* Wo        = (const float*)d_in[10];
    const float* bo        = (const float*)d_in[11];
    const float* pe_q      = (const float*)d_in[12];
    const float* pe_k      = (const float*)d_in[13];
    const float* cp        = (const float*)d_in[14];
    const float* W_off     = (const float*)d_in[15];

    const int NQ = in_sizes[0] / 9;                        // bs*ns
    const int bs = in_sizes[1] / (C_DIM * HP * HP);
    const int ns = NQ / bs;
    const size_t NQs = (size_t)NQ;

    float* w      = (float*)d_ws;
    float* feat   = w;  w += NQs * C_DIM;
    float* pts    = w;  w += NQs * NCP * 3;
    float* q_ws   = w;  w += NQs * C_DIM;
    float* qk_ws  = w;  w += NQs * C_DIM;
    float* sbar   = w;  w += NQs * C_DIM;

    ushort_t* us      = (ushort_t*)w;
    ushort_t* feat_bf = us;  us += NQs * C_DIM;
    ushort_t* sbar_bf = us;  us += NQs * C_DIM;
    ushort_t* Wq_p    = us;  us += WPACK;
    ushort_t* Wk_p    = us;  us += WPACK;
    ushort_t* Wv_p    = us;  us += WPACK;
    ushort_t* Wo_p    = us;  us += WPACK;

    float* out = (float*)d_out;

    // Pack weights into fragment-ordered bf16.
    pack_w_kernel<<<WPACK / 256, 256, 0, stream>>>(Wq, Wq_p, 1);   // X @ Wq^T
    pack_w_kernel<<<WPACK / 256, 256, 0, stream>>>(Wk, Wk_p, 0);   // X @ Wk
    pack_w_kernel<<<WPACK / 256, 256, 0, stream>>>(Wv, Wv_p, 1);   // X @ Wv^T
    pack_w_kernel<<<WPACK / 256, 256, 0, stream>>>(Wo, Wo_p, 1);   // X @ Wo^T

    geom_kernel<<<NQ, C_DIM, 0, stream>>>(query_pos, c_xy, c_xz, c_yz, cp, W_off,
                                          feat, feat_bf, pts, ns);

    int gemm_blocks = (NQ / 16 + 3) / 4;       // 4 wave-tiles (16 rows) per block
    gemm_q_qk_kernel<<<gemm_blocks, 128, 0, stream>>>(feat_bf, Wq_p, Wk_p,
                                                      bq, pe_q, q_ws, qk_ws, NQ);

    attn_kernel<<<NQ, C_DIM, 0, stream>>>(pts, q_ws, qk_ws, bk, pe_k,
                                          c_xy, c_xz, c_yz, sbar, sbar_bf, ns);

    gemm_out_kernel<<<gemm_blocks, 128, 0, stream>>>(sbar_bf, Wv_p, Wo_p,
                                                     bv, bo, feat, out, NQ);
}
